// DecoderRNN_12945031431021
// MI455X (gfx1250) — compile-verified, matrix operand order checked by reading, not run on hardware
//
#include <hip/hip_runtime.h>
#include <math.h>

// ---------------------------------------------------------------------------
// DecoderRNN (LSTM, B=128 T=24 H=E=1024 V=32000) for gfx1250 / MI455X.
// GEMMs on v_wmma_f32_16x16x32_bf16 (wave32). The shared/skinny A operand
// (h or x_t) is staged through a double-buffered 2x32KB LDS pipeline:
// while the 32-WMMA chain of chunk kc runs (A frags via ds_load_b128), the
// global loads for chunk kc+1 are already in flight into registers, stored
// to the other LDS buffer after the WMMAs (one barrier per chunk). The
// streaming B operand (bf16 weights, L2-resident) is pipelined one K-step
// ahead. Output stores are non-temporal to keep W_out hot in the 192MB L2.
// ---------------------------------------------------------------------------

typedef __attribute__((ext_vector_type(16))) __bf16       v16bf;
typedef __attribute__((ext_vector_type(8)))  float        v8f;
typedef __attribute__((ext_vector_type(4)))  unsigned int u32x4;

#define B_   128
#define T_   24
#define H_   1024
#define E_   1024
#define V_   32000
#define G4_  4096                 // 4*H
#define KC_  128                  // K-chunk in LDS (128 x 128 bf16 = 32KB)
#define CQ_  (128 * (KC_ >> 3))   // u32x4 per chunk = 2048
#define NQ_  (CQ_ / 256)          // u32x4 per thread per chunk = 8

union FragBF { v16bf v; u32x4 q[2]; };

__device__ __forceinline__ __bf16 f2bf(float f) {
  unsigned u = __float_as_uint(f);
  unsigned r = (u + 0x7FFFu + ((u >> 16) & 1u)) >> 16;   // round-to-nearest-even
  unsigned short s = (unsigned short)r;
  __bf16 b;
  __builtin_memcpy(&b, &s, 2);
  return b;
}

// 16x32 bf16 fragment gather per the CDNA5 16-bit A/B layout:
//   lanes 0-15 hold row (lane&15), K in {k0..k0+7, k0+16..k0+23}
//   lanes 16-31 hold row (lane&15), K in {k0+8..k0+15, k0+24..k0+31}
// Two b128 loads per lane. Serves the B operand of A @ W^T when W is stored
// row-major [N,K] (B[k][n] == W[n][k]).
__device__ __forceinline__ v16bf load_frag_g(const __bf16* __restrict__ base,
                                             int ld, int row0, int k0, int lane) {
  const int r  = lane & 15;
  const int ks = (lane >> 4) << 3;            // 0 or 8
  const __bf16* p = base + (size_t)(row0 + r) * ld + (k0 + ks);
  FragBF f;
  f.q[0] = *(const u32x4*)(p);
  f.q[1] = *(const u32x4*)(p + 16);
  return f.v;
}

// Same gather from an LDS-staged [128 x KC_] chunk -> ds_load_b128 x2.
__device__ __forceinline__ v16bf load_frag_lds(const __bf16* base,
                                               int row0, int k0, int lane) {
  const int r  = lane & 15;
  const int ks = (lane >> 4) << 3;
  const __bf16* p = base + (row0 + r) * KC_ + (k0 + ks);
  FragBF f;
  f.q[0] = *(const u32x4*)(p);
  f.q[1] = *(const u32x4*)(p + 16);
  return f.v;
}

// Batched global loads of A[:, kc:kc+KC_] (row-major [128, ld] bf16) into
// registers: NQ_ b128 loads per thread issued back-to-back (one clause),
// so their latency overlaps the WMMA chain of the current chunk.
__device__ __forceinline__ void stage_load(u32x4 (&tmp)[NQ_],
                                           const __bf16* __restrict__ src,
                                           int ld, int kc) {
  const u32x4* s = (const u32x4*)src;
  const int row_q = ld >> 3;
#pragma unroll
  for (int i = 0; i < NQ_; ++i) {
    const int idx = threadIdx.x + i * 256;          // 0..CQ_-1, coalesced
    const int row = idx / (KC_ >> 3);
    const int col = idx % (KC_ >> 3);
    tmp[i] = s[(size_t)row * row_q + (kc >> 3) + col];
  }
}

__device__ __forceinline__ void stage_store(__bf16* dst, const u32x4 (&tmp)[NQ_]) {
  u32x4* d = (u32x4*)dst;
#pragma unroll
  for (int i = 0; i < NQ_; ++i) d[threadIdx.x + i * 256] = tmp[i];
}

#define WMMA_BF16(Afrag, Bfrag, Cacc) \
  __builtin_amdgcn_wmma_f32_16x16x32_bf16(false, (Afrag), false, (Bfrag), \
                                          (short)0, (Cacc), false, false)

// B-fragment for the gates GEMM viewed as one virtual K=2048 reduction:
// k < 1024 -> W_ih, else W_hh (column k & 1023).
__device__ __forceinline__ v16bf gate_bfrag(const __bf16* __restrict__ Wih,
                                            const __bf16* __restrict__ Whh,
                                            int n0, int k, int lane) {
  const __bf16* W = (k < H_) ? Wih : Whh;
  return load_frag_g(W, H_, n0, k & (H_ - 1), lane);
}

// ---------------------------------------------------------------------------
// gates = x @ W_ih^T + h @ W_hh^T + (b_ih + b_hh)    -> [128, 4096] fp32
// grid = 32 WGs x 256 thr; wave w owns N strip [blk*128+16w,+16) over all
// 128 M rows (8 accum tiles of 16x16).
// ---------------------------------------------------------------------------
__global__ __launch_bounds__(256) void gates_kernel(
    const __bf16* __restrict__ x,    // [128,1024] bf16
    const __bf16* __restrict__ h,    // [128,1024] bf16
    const __bf16* __restrict__ Wih,  // [4096,1024] bf16 row-major
    const __bf16* __restrict__ Whh,  // [4096,1024] bf16 row-major
    const float*  __restrict__ b_ih,
    const float*  __restrict__ b_hh,
    float*        __restrict__ gates) {
  __shared__ __bf16 smem[2][128 * KC_];       // 2 x 32KB double buffer
  const int lane = threadIdx.x & 31;
  const int wave = threadIdx.x >> 5;
  const int n0   = blockIdx.x * 128 + wave * 16;

  v8f   acc[8] = {};
  u32x4 tmp[NQ_];

  stage_load(tmp, x, H_, 0);                  // prologue: chunk 0 of x
  stage_store(smem[0], tmp);
  __syncthreads();
  v16bf bcur = gate_bfrag(Wih, Whh, n0, 0, lane);

  for (int kc = 0; kc < 2 * H_; kc += KC_) {
    const int  buf      = (kc / KC_) & 1;
    const bool has_next = (kc + KC_ < 2 * H_);
    if (has_next) {
      const int kn = kc + KC_;
      stage_load(tmp, (kn < H_) ? x : h, H_, kn & (H_ - 1));  // in flight
    }
    for (int k0 = 0; k0 < KC_; k0 += 32) {
      const int knext = kc + k0 + 32;
      v16bf bnxt = (knext < 2 * H_) ? gate_bfrag(Wih, Whh, n0, knext, lane)
                                    : bcur;
#pragma unroll
      for (int mt = 0; mt < 8; ++mt) {
        v16bf af = load_frag_lds(smem[buf], mt * 16, k0, lane);
        acc[mt] = WMMA_BF16(af, bcur, acc[mt]);
      }
      bcur = bnxt;
    }
    if (has_next) stage_store(smem[buf ^ 1], tmp);
    __syncthreads();
  }

  const int   n    = n0 + (lane & 15);
  const int   mr   = (lane >> 4) << 3;        // 0 or 8
  const float bias = b_ih[n] + b_hh[n];
#pragma unroll
  for (int mt = 0; mt < 8; ++mt)
#pragma unroll
    for (int r = 0; r < 8; ++r)
      gates[(size_t)(mt * 16 + mr + r) * G4_ + n] = acc[mt][r] + bias;
}

// ---------------------------------------------------------------------------
// logits = h @ W_out^T + b_out -> out[:, t+1, :]   (grid = 250 WGs)
// W_out streams from L2 exactly once per wave (pipelined); h comes from LDS.
// Output stores are non-temporal (write-once, never read).
// ---------------------------------------------------------------------------
__global__ __launch_bounds__(256) void logits_kernel(
    const __bf16* __restrict__ h,     // [128,1024] bf16
    const __bf16* __restrict__ Wout,  // [32000,1024] bf16 row-major
    const float*  __restrict__ b_out,
    float*        __restrict__ out,   // [128, 25, 32000] fp32
    int t) {
  __shared__ __bf16 smem[2][128 * KC_];       // 2 x 32KB double buffer
  const int lane = threadIdx.x & 31;
  const int wave = threadIdx.x >> 5;
  const int n0   = blockIdx.x * 128 + wave * 16;

  v8f   acc[8] = {};
  u32x4 tmp[NQ_];

  stage_load(tmp, h, H_, 0);
  stage_store(smem[0], tmp);
  __syncthreads();
  v16bf bcur = load_frag_g(Wout, H_, n0, 0, lane);

  for (int kc = 0; kc < H_; kc += KC_) {
    const int  buf      = (kc / KC_) & 1;
    const bool has_next = (kc + KC_ < H_);
    if (has_next) stage_load(tmp, h, H_, kc + KC_);           // in flight
    for (int k0 = 0; k0 < KC_; k0 += 32) {
      const int knext = kc + k0 + 32;
      v16bf bnxt = (knext < H_) ? load_frag_g(Wout, H_, n0, knext, lane)
                                : bcur;
#pragma unroll
      for (int mt = 0; mt < 8; ++mt) {
        v16bf af = load_frag_lds(smem[buf], mt * 16, k0, lane);
        acc[mt] = WMMA_BF16(af, bcur, acc[mt]);
      }
      bcur = bnxt;
    }
    if (has_next) stage_store(smem[buf ^ 1], tmp);
    __syncthreads();
  }

  const int    n    = n0 + (lane & 15);
  const int    mr   = (lane >> 4) << 3;
  const float  bias = b_out[n];
  const size_t rs   = (size_t)(T_ + 1) * V_;
  float* o = out + (size_t)(t + 1) * V_ + n;
#pragma unroll
  for (int mt = 0; mt < 8; ++mt)
#pragma unroll
    for (int r = 0; r < 8; ++r)
      __builtin_nontemporal_store(acc[mt][r] + bias,
                                  o + (size_t)(mt * 16 + mr + r) * rs);
}

// ---------------------------------------------------------------------------
// Elementwise LSTM cell: c = sig(f)*c + sig(i)*tanh(g); h = sig(o)*tanh(c)
// ---------------------------------------------------------------------------
__global__ void update_kernel(const float* __restrict__ gates,
                              float* __restrict__ c, __bf16* __restrict__ h) {
  const int j = blockIdx.x * blockDim.x + threadIdx.x;
  if (j >= B_ * H_) return;
  const int b = j >> 10, u = j & (H_ - 1);
  const float* g = gates + (size_t)b * G4_;
  const float gi = g[u], gf = g[H_ + u], gg = g[2 * H_ + u], go = g[3 * H_ + u];
  const float si = 1.0f / (1.0f + expf(-gi));
  const float sf = 1.0f / (1.0f + expf(-gf));
  const float so = 1.0f / (1.0f + expf(-go));
  const float cn = sf * c[j] + si * tanhf(gg);
  c[j] = cn;
  h[j] = f2bf(so * tanhf(cn));
}

// x_t = bf16(emb[captions[:, t]])    [128,1024]
__global__ void gather_kernel(const float* __restrict__ emb,
                              const int* __restrict__ captions, int t,
                              __bf16* __restrict__ x) {
  const int j = blockIdx.x * blockDim.x + threadIdx.x;
  if (j >= B_ * E_) return;
  const int b = j >> 10, e = j & (E_ - 1);
  const int tok = captions[b * T_ + t];
  x[j] = f2bf(emb[(size_t)tok * E_ + e]);
}

// fp32 -> bf16 weight conversion, grid-stride
__global__ void cvt_kernel(const float* __restrict__ src,
                           __bf16* __restrict__ dst, long n) {
  for (long i = blockIdx.x * (long)blockDim.x + threadIdx.x; i < n;
       i += (long)gridDim.x * blockDim.x)
    dst[i] = f2bf(src[i]);
}

// h = c = features (h kept in bf16 for the GEMMs, c in fp32)
__global__ void init_state_kernel(const float* __restrict__ features,
                                  float* __restrict__ c, __bf16* __restrict__ h) {
  const int j = blockIdx.x * blockDim.x + threadIdx.x;
  if (j >= B_ * H_) return;
  const float v = features[j];
  c[j] = v;
  h[j] = f2bf(v);
}

// out[:, 0, :] = one-hot(1)
__global__ void start_kernel(float* __restrict__ out) {
  const long j = blockIdx.x * (long)blockDim.x + threadIdx.x;
  if (j >= (long)B_ * V_) return;
  const long b = j / V_, v = j % V_;
  __builtin_nontemporal_store((v == 1) ? 1.0f : 0.0f,
                              out + b * (long)(T_ + 1) * V_ + v);
}

// ---------------------------------------------------------------------------
extern "C" void kernel_launch(void* const* d_in, const int* in_sizes, int n_in,
                              void* d_out, int out_size, void* d_ws, size_t ws_size,
                              hipStream_t stream) {
  (void)in_sizes; (void)n_in; (void)out_size; (void)ws_size;

  const float* features = (const float*)d_in[0];   // [128,1024]
  const int*   captions = (const int*)  d_in[1];   // [128,24]
  const float* emb      = (const float*)d_in[2];   // [32000,1024]
  const float* W_ih     = (const float*)d_in[3];   // [4096,1024]
  const float* W_hh     = (const float*)d_in[4];   // [4096,1024]
  const float* b_ih     = (const float*)d_in[5];   // [4096]
  const float* b_hh     = (const float*)d_in[6];   // [4096]
  const float* W_out    = (const float*)d_in[7];   // [32000,1024]
  const float* b_out    = (const float*)d_in[8];   // [32000]
  float* out = (float*)d_out;                      // [128,25,32000]

  // Workspace layout (bytes)
  char* ws = (char*)d_ws;
  __bf16* Wih_bf  = (__bf16*)(ws + 0);                       //  8 MiB
  __bf16* Whh_bf  = (__bf16*)(ws + (size_t)8  * 1024 * 1024);//  8 MiB
  __bf16* Wout_bf = (__bf16*)(ws + (size_t)16 * 1024 * 1024);// 62.5 MiB
  size_t  off     = (size_t)16 * 1024 * 1024 + (size_t)V_ * H_ * 2;
  float*  c_st    = (float*)(ws + off);  off += (size_t)B_ * H_ * 4;
  __bf16* h_bf    = (__bf16*)(ws + off); off += (size_t)B_ * H_ * 2;
  __bf16* x_bf    = (__bf16*)(ws + off); off += (size_t)B_ * E_ * 2;
  float*  gates   = (float*)(ws + off);  // 2 MiB

  // 1) Weights -> bf16 (stateless: redo every call)
  cvt_kernel<<<4096, 256, 0, stream>>>(W_ih,  Wih_bf,  (long)G4_ * H_);
  cvt_kernel<<<4096, 256, 0, stream>>>(W_hh,  Whh_bf,  (long)G4_ * H_);
  cvt_kernel<<<8192, 256, 0, stream>>>(W_out, Wout_bf, (long)V_  * H_);

  // 2) State init + start-token row
  init_state_kernel<<<(B_ * H_ + 255) / 256, 256, 0, stream>>>(features, c_st, h_bf);
  start_kernel<<<((long)B_ * V_ + 255) / 256, 256, 0, stream>>>(out);

  // 3) Sequential LSTM steps
  for (int t = 0; t < T_; ++t) {
    gather_kernel<<<(B_ * E_ + 255) / 256, 256, 0, stream>>>(emb, captions, t, x_bf);
    gates_kernel<<<G4_ / 128, 256, 0, stream>>>(x_bf, h_bf, Wih_bf, Whh_bf,
                                                b_ih, b_hh, gates);
    update_kernel<<<(B_ * H_ + 255) / 256, 256, 0, stream>>>(gates, c_st, h_bf);
    logits_kernel<<<V_ / 128, 256, 0, stream>>>(h_bf, Wout_bf, b_out, out, t);
  }
}
